// FullRelPos_30983894073933
// MI455X (gfx1250) — matrix operand order repaired and checked by reading, not compiled, not run on hardware
//
#include <hip/hip_runtime.h>

typedef __attribute__((ext_vector_type(16))) _Float16 v16h;
typedef __attribute__((ext_vector_type(8)))  float    v8f;
typedef __attribute__((ext_vector_type(4)))  float    v4f;

namespace {
constexpr int kH = 32, kW = 32, kG = 8, kDim = 64, kC = 32;
constexpr int kBlockElems = kG * kH * kW;          // 8192 attn floats per (b,h,w)
constexpr int kVec4PerBlk = kBlockElems / 4;       // 2048 float4s
}

// One block per (b, h, w): waves 0/1 do the relative-position GEMMs with WMMA,
// then all 8 waves stream attn -> out with the broadcast add.
__global__ __launch_bounds__(256)
void fullrelpos_fused(const float* __restrict__ q,
                      const float* __restrict__ attn,
                      const float* __restrict__ rel_h,
                      const float* __restrict__ rel_w,
                      float* __restrict__ out)
{
    __shared__ float s_lh[kG * kH];   // logits_h[g][kh]
    __shared__ float s_lw[kG * kW];   // logits_w[g][kw]

    const int blk = blockIdx.x;       // b*H*W + h*W + w
    const int h   = (blk >> 5) & 31;
    const int w   = blk & 31;

    const int tid  = threadIdx.x;
    const int wave = tid >> 5;
    const int lane = tid & 31;

    if (wave < 2) {
        // wave 0: logits_h (q channel-half 0, rel_emb_h, position h)
        // wave 1: logits_w (q channel-half 1, rel_emb_w, position w)
        const float* rel   = (wave == 0) ? rel_h : rel_w;
        const int    pos   = (wave == 0) ? h : w;
        const int    chalf = wave * kC;
        float*       sdst  = (wave == 0) ? s_lh : s_lw;

        const int col = lane & 15;    // A: row m; B: column n; D: N index
        const int hi  = lane >> 4;    // lane half selects K sub-range

        // ---- A operand (16x32 f16 layout, ISA 7.12.2): row m = g, m>=8 zero.
        // lanes 0-15: elems 0..7 = K0..7,  8..15 = K16..23
        // lanes 16-31: elems 0..7 = K8..15, 8..15 = K24..31
        // Split-f16: x = hi + lo for near-fp32 accuracy with f32 accumulate.
        v16h a_hi = {};
        v16h a_lo = {};
        if (col < kG) {
            const float* qrow = q + ((size_t)blk * kG + col) * kDim + chalf;
            const int klo = hi ? 8 : 0;
            #pragma unroll
            for (int i = 0; i < 8; ++i) {
                float x0 = qrow[klo + i];
                float x1 = qrow[16 + klo + i];
                _Float16 h0 = (_Float16)x0;
                _Float16 h1 = (_Float16)x1;
                a_hi[i]     = h0;
                a_hi[8 + i] = h1;
                a_lo[i]     = (_Float16)(x0 - (float)h0);
                a_lo[8 + i] = (_Float16)(x1 - (float)h1);
            }
        }

        #pragma unroll
        for (int t = 0; t < 2; ++t) {             // two N tiles cover k = 0..31
            // ---- B operand (32x16 f16): lane holds column n = col,
            // elems 0..7 = K hi*8..hi*8+7, 8..15 = K 16+hi*8..23+hi*8. B[k][n] = rel[(n + 31 - pos)*32 + k].
            const int n   = col + 16 * t;
            const int row = n + 31 - pos;         // in [0, 62]
            const float* brow = rel + (size_t)row * kC + hi * 8;
            v16h b_hi = {};
            v16h b_lo = {};
            #pragma unroll
            for (int i = 0; i < 16; ++i) {
                float x = brow[(i & 7) + ((i >> 3) << 4)];
                _Float16 hh = (_Float16)x;
                b_hi[i] = hh;
                b_lo[i] = (_Float16)(x - (float)hh);
            }

            v8f acc = {};
            acc = __builtin_amdgcn_wmma_f32_16x16x32_f16(false, a_hi, false, b_hi,
                                                         (short)0, acc, false, false);
            acc = __builtin_amdgcn_wmma_f32_16x16x32_f16(false, a_hi, false, b_lo,
                                                         (short)0, acc, false, false);
            acc = __builtin_amdgcn_wmma_f32_16x16x32_f16(false, a_lo, false, b_hi,
                                                         (short)0, acc, false, false);

            // D layout: VGPR r / lanes 0-15 -> (M=r, N=lane). Only M<8 (g) valid.
            if (hi == 0) {
                #pragma unroll
                for (int r = 0; r < kG; ++r)
                    sdst[r * 32 + 16 * t + col] = acc[r];
            }
        }
    }
    __syncthreads();

    // ---- Streaming phase: out = attn + lh[g,kh] + lw[g,kw], 128-bit NT I/O.
    const v4f* attn4 = (const v4f*)attn + (size_t)blk * kVec4PerBlk;
    v4f*       out4  = (v4f*)out        + (size_t)blk * kVec4PerBlk;
    const v4f* lw4   = (const v4f*)s_lw;

    #pragma unroll
    for (int j = 0; j < kVec4PerBlk / 256; ++j) {   // 8 iters
        const int i4 = tid + j * 256;               // float4 index in block slab
        const int e  = i4 << 2;                     // element index: g*1024+kh*32+kw
        const int g  = e >> 10;
        const int kh = (e >> 5) & 31;
        v4f   av  = __builtin_nontemporal_load(attn4 + i4);
        float lhv = s_lh[g * 32 + kh];
        v4f   lwv = lw4[(g * 32 + (e & 31)) >> 2];
        v4f   r   = av + lhv + lwv;
        __builtin_nontemporal_store(r, out4 + i4);
    }
}

extern "C" void kernel_launch(void* const* d_in, const int* in_sizes, int n_in,
                              void* d_out, int out_size, void* d_ws, size_t ws_size,
                              hipStream_t stream) {
    (void)n_in; (void)out_size; (void)d_ws; (void)ws_size;
    const float* q     = (const float*)d_in[0];
    const float* attn  = (const float*)d_in[1];
    const float* rel_h = (const float*)d_in[2];
    const float* rel_w = (const float*)d_in[3];
    float* out = (float*)d_out;
    const int nblk = in_sizes[1] / kBlockElems;     // B * H * W workgroups
    fullrelpos_fused<<<nblk, 256, 0, stream>>>(q, attn, rel_h, rel_w, out);
}
